// CausalSelfAttention_69569880260840
// MI455X (gfx1250) — matrix operand order repaired and checked by reading, not compiled, not run on hardware
//
#include <hip/hip_runtime.h>

typedef __attribute__((ext_vector_type(16))) __bf16 v16bf;
typedef __attribute__((ext_vector_type(8)))  __bf16 v8bf;
typedef __attribute__((ext_vector_type(8)))  float  v8f;
typedef __attribute__((ext_vector_type(4)))  unsigned int u32x4;
typedef __attribute__((ext_vector_type(8)))  int i32x8;
typedef __attribute__((ext_vector_type(4)))  int i32x4;

#if defined(__has_builtin)
#if __has_builtin(__builtin_amdgcn_tensor_load_to_lds)
#define HAVE_TDM 1
#endif
#endif
#ifndef HAVE_TDM
#define HAVE_TDM 0
#endif

__device__ __forceinline__ v16bf cat16(v8bf lo, v8bf hi) {
  v16bf r;
#pragma unroll
  for (int i = 0; i < 8; ++i) { r[i] = lo[i]; r[i + 8] = hi[i]; }
  return r;
}

__device__ __forceinline__ v8f wmma_bf16(v16bf a, v16bf b, v8f c) {
  return __builtin_amdgcn_wmma_f32_16x16x32_bf16(false, a, false, b, (short)0, c,
                                                 false, false);
}

// CDNA5 async global<->LDS (ASYNCcnt tracked), 16B per lane.
__device__ __forceinline__ void async_copy_b128(void* lds_dst, const void* gsrc) {
  const unsigned int lds_off = (unsigned int)(uintptr_t)lds_dst;
  asm volatile("global_load_async_to_lds_b128 %0, %1, off"
               :: "v"(lds_off), "v"(gsrc) : "memory");
}
__device__ __forceinline__ void async_store_b128(void* gdst, const void* lds_src) {
  const unsigned int lds_off = (unsigned int)(uintptr_t)lds_src;
  asm volatile("global_store_async_from_lds_b128 %0, %1, off"
               :: "v"(gdst), "v"(lds_off) : "memory");
}
__device__ __forceinline__ void wait_asynccnt0() {
  asm volatile("s_wait_asynccnt 0x0" ::: "memory");
}

#if HAVE_TDM
// TDM: 2D bf16 tile (rows x width elems, global row stride) -> LDS,
// optional LDS row padding (pad_amount DWORD-code after every interval code).
__device__ __forceinline__ void tdm_load_2d_bf16(const void* gsrc, unsigned lds_addr,
                                                 unsigned width, unsigned rows,
                                                 unsigned row_stride,
                                                 unsigned pad_int_code,
                                                 unsigned pad_amt_code) {
  const unsigned long long ga = (unsigned long long)(uintptr_t)gsrc;
  u32x4 g0;
  g0[0] = 1u;                                   // count=1 valid user descriptor
  g0[1] = lds_addr;                             // LDS byte address
  g0[2] = (unsigned)(ga & 0xffffffffu);         // global_addr[31:0]
  g0[3] = (unsigned)((ga >> 32) & 0x01ffffffu)  // global_addr[56:32]
          | (2u << 30);                         // type = 2 ("image")
  i32x8 g1;
  g1[0] = (int)((1u << 16)                      // data_size = 1 -> 2 bytes
                | (1u << 20)                    // pad_enable
                | (pad_int_code << 22) | (pad_amt_code << 25));
  g1[1] = (int)((width & 0xffffu) << 16);                         // dim0[15:0]
  g1[2] = (int)(((width >> 16) & 0xffffu) | ((rows & 0xffffu) << 16)); // dim0hi|dim1lo
  g1[3] = (int)(((rows >> 16) & 0xffffu) | ((width & 0xffffu) << 16)); // dim1hi|tile0
  g1[4] = (int)(rows & 0xffffu);                                  // tile1, tile2=0
  g1[5] = (int)row_stride;                                        // stride0[31:0]
  g1[6] = 0;
  g1[7] = 0;
  i32x4 z4; z4[0] = z4[1] = z4[2] = z4[3] = 0;
#if __clang_major__ >= 23
  i32x8 z8;
#pragma unroll
  for (int i = 0; i < 8; ++i) z8[i] = 0;
  __builtin_amdgcn_tensor_load_to_lds(g0, g1, z4, z4, z8, 0);
#else
  __builtin_amdgcn_tensor_load_to_lds(g0, g1, z4, z4, 0);
#endif
}
#endif

// ---------------------------------------------------------------------------
// GEMM: Out[M,N] = A[M,K] @ W[K,N] + bias[N]
// Block tile 128x128, 8 waves x (32x64), K-step 64, register-prefetched
// staging + global_prefetch two tiles ahead.
// ---------------------------------------------------------------------------
template <typename AT, typename OT>
__global__ __launch_bounds__(256)
void gemm_bf16_wmma_kernel(const AT* __restrict__ A, const float* __restrict__ W,
                           const float* __restrict__ bias, OT* __restrict__ Out,
                           int M, int N, int K) {
  constexpr int KS  = 64;
  constexpr int PAD = 72;
  __shared__ alignas(16) __bf16 As[128 * PAD];  // [m][k]
  __shared__ alignas(16) __bf16 Ws[128 * PAD];  // transposed: [n][k]

  const int tid  = threadIdx.x;
  const int lane = tid & 31;
  const int wave = tid >> 5;
  const int r    = lane & 15;
  const int hi   = lane >> 4;

  const int bm = blockIdx.y * 128;
  const int bn = blockIdx.x * 128;
  const int wm = (wave >> 1) * 32;
  const int wn = (wave & 1) * 64;

  const int ar = tid >> 1, ko = (tid & 1) * 32;
  const int wc = tid & 127, kh = (tid >> 7) * 32;

  v8f acc[2][4];
#pragma unroll
  for (int mt = 0; mt < 2; ++mt)
#pragma unroll
    for (int nt = 0; nt < 4; ++nt)
#pragma unroll
      for (int v = 0; v < 8; ++v) acc[mt][nt][v] = 0.0f;

  float aReg[32], wReg[32];
  const AT* gA = A + (size_t)(bm + ar) * K + ko;
  const float* gW = W + (size_t)kh * N + bn + wc;

#pragma unroll
  for (int i = 0; i < 32; ++i) aReg[i] = (float)gA[i];
#pragma unroll
  for (int i = 0; i < 32; ++i) wReg[i] = gW[(size_t)i * N];

  for (int kk = 0; kk < K; kk += KS) {
#pragma unroll
    for (int j = 0; j < 4; ++j) {
      v8bf t;
#pragma unroll
      for (int i = 0; i < 8; ++i) t[i] = (__bf16)aReg[j * 8 + i];
      *(v8bf*)&As[ar * PAD + ko + j * 8] = t;
    }
#pragma unroll
    for (int j = 0; j < 4; ++j) {
      v8bf t;
#pragma unroll
      for (int i = 0; i < 8; ++i) t[i] = (__bf16)wReg[j * 8 + i];
      *(v8bf*)&Ws[wc * PAD + kh + j * 8] = t;
    }
    __syncthreads();

    if (kk + KS < K) {
      if (kk + 2 * KS < K) {
        __builtin_prefetch(gA + (kk + 2 * KS), 0, 1);
        __builtin_prefetch(gW + (size_t)(kk + 2 * KS) * N, 0, 1);
      }
#pragma unroll
      for (int i = 0; i < 32; ++i) aReg[i] = (float)gA[kk + KS + i];
#pragma unroll
      for (int i = 0; i < 32; ++i) wReg[i] = gW[(size_t)(kk + KS + i) * N];
    }

#pragma unroll
    for (int kc = 0; kc < 2; ++kc) {
      v16bf af[2], bfrag[4];
#pragma unroll
      for (int mt = 0; mt < 2; ++mt) {
        const __bf16* p = &As[(wm + mt * 16 + r) * PAD + kc * 32 + hi * 8];
        af[mt] = cat16(*(const v8bf*)p, *(const v8bf*)(p + 16));
      }
#pragma unroll
      for (int nt = 0; nt < 4; ++nt) {
        const __bf16* p = &Ws[(wn + nt * 16 + r) * PAD + kc * 32 + hi * 8];
        bfrag[nt] = cat16(*(const v8bf*)p, *(const v8bf*)(p + 16));
      }
#pragma unroll
      for (int mt = 0; mt < 2; ++mt)
#pragma unroll
        for (int nt = 0; nt < 4; ++nt)
          acc[mt][nt] = wmma_bf16(af[mt], bfrag[nt], acc[mt][nt]);
    }
    __syncthreads();
  }

#pragma unroll
  for (int mt = 0; mt < 2; ++mt)
#pragma unroll
    for (int nt = 0; nt < 4; ++nt) {
      const int gn = bn + wn + nt * 16 + r;
      const float bv = bias[gn];
#pragma unroll
      for (int v = 0; v < 8; ++v) {
        const int gm = bm + wm + mt * 16 + v + 8 * hi;
        Out[(size_t)gm * N + gn] = (OT)(acc[mt][nt][v] + bv);
      }
    }
}

// ---------------------------------------------------------------------------
// Flash attention. qkv: bf16 [B][T][3C]; Y: bf16 [B][T][C].
// K tile via TDM (one 2D descriptor, padded LDS pitch, TENSORcnt) with async
// fallback; V transposed via VGPRs; Y epilogue via async LDS->global stores.
// ---------------------------------------------------------------------------
__global__ __launch_bounds__(256)
void attn_fwd_kernel(const __bf16* __restrict__ qkv, __bf16* __restrict__ Y,
                     int B, int H, int T, int C) {
  constexpr int D = 128;
  constexpr int KPAD = 136;  // 272B LDS pitch (= 256B row + 16B TDM pad)
  constexpr int VPAD = 40;
  constexpr int PPAD = 40;
  __shared__ alignas(16) __bf16 smem[16384];  // 32KB arena
  __bf16* Ksm = smem;               // [32][KPAD]   (4352 el)
  __bf16* Vsm = smem + 4352;        // [D][VPAD]    (5120 el)
  __bf16* Psm = smem + 9472;        // 8 x [16][PPAD] (5120 el)

  const int tid = threadIdx.x, lane = tid & 31, wave = tid >> 5;
  const int r = lane & 15, hi = lane >> 4;
  const int qTiles = T / 128;
  const int bid = blockIdx.x;
  const int qb = (bid % qTiles) * 128;
  const int h  = (bid / qTiles) % H;
  const int b  = bid / (qTiles * H);

  const size_t rs = (size_t)3 * C;
  const __bf16* Qb = qkv + (size_t)b * T * rs + (size_t)h * D;
  const __bf16* Kb = Qb + C;
  const __bf16* Vb = Qb + 2 * C;

  const float scale = 0.08838834764831845f;  // 1/sqrt(128)

  const int qrow = qb + wave * 16 + r;
  v16bf qf[4];
#pragma unroll
  for (int c = 0; c < 4; ++c) {
    const __bf16* p = Qb + (size_t)qrow * rs + c * 32 + hi * 8;
    qf[c] = cat16(*(const v8bf*)p, *(const v8bf*)(p + 16));
  }

  float mrow[8], lrow[8];
  v8f acc[8];
#pragma unroll
  for (int v = 0; v < 8; ++v) { mrow[v] = -3.0e38f; lrow[v] = 0.0f; }
#pragma unroll
  for (int dt = 0; dt < 8; ++dt)
#pragma unroll
    for (int v = 0; v < 8; ++v) acc[dt][v] = 0.0f;

  const int key  = tid >> 3;
  const int doff = (tid & 7) * 16;

  const int kEnd = qb + 128;
  for (int kb = 0; kb < kEnd; kb += 32) {
#if HAVE_TDM
    if (wave == 0) {  // one descriptor moves the whole 32x256B K tile
      tdm_load_2d_bf16(Kb + (size_t)kb * rs, (unsigned)(uintptr_t)Ksm,
                       /*width=*/D, /*rows=*/32, /*row_stride=*/(unsigned)rs,
                       /*pad_int(64dw)=*/5u, /*pad_amt(4dw)=*/3u);
    }
#else
    {
      const __bf16* kp = Kb + (size_t)(kb + key) * rs + doff;
      async_copy_b128(&Ksm[key * KPAD + doff], kp);
      async_copy_b128(&Ksm[key * KPAD + doff + 8], kp + 8);
    }
#endif
    {  // V tile: needs transpose -> VGPR path
      const __bf16* vp = Vb + (size_t)(kb + key) * rs + doff;
      v8bf v0 = *(const v8bf*)vp;
      v8bf v1 = *(const v8bf*)(vp + 8);
#pragma unroll
      for (int i = 0; i < 8; ++i) Vsm[(doff + i) * VPAD + key] = v0[i];
#pragma unroll
      for (int i = 0; i < 8; ++i) Vsm[(doff + 8 + i) * VPAD + key] = v1[i];
    }
#if HAVE_TDM
    __builtin_amdgcn_s_wait_tensorcnt(0);
#else
    wait_asynccnt0();
#endif
    __syncthreads();

    // S = Q @ K^T
    v8f s0, s1;
#pragma unroll
    for (int v = 0; v < 8; ++v) { s0[v] = 0.0f; s1[v] = 0.0f; }
#pragma unroll
    for (int c = 0; c < 4; ++c) {
      const __bf16* p0 = &Ksm[r * KPAD + c * 32 + hi * 8];
      const __bf16* p1 = &Ksm[(16 + r) * KPAD + c * 32 + hi * 8];
      s0 = wmma_bf16(qf[c], cat16(*(const v8bf*)p0, *(const v8bf*)(p0 + 16)), s0);
      s1 = wmma_bf16(qf[c], cat16(*(const v8bf*)p1, *(const v8bf*)(p1 + 16)), s1);
    }

    // scale + causal mask + online softmax
    const int col0 = kb + r, col1 = kb + 16 + r;
#pragma unroll
    for (int v = 0; v < 8; ++v) {
      const int row = qb + wave * 16 + v + 8 * hi;
      float a0 = s0[v] * scale;
      float a1 = s1[v] * scale;
      if (col0 > row) a0 = -1.0e30f;
      if (col1 > row) a1 = -1.0e30f;

      float t = fmaxf(a0, a1);
#pragma unroll
      for (int off = 8; off >= 1; off >>= 1) t = fmaxf(t, __shfl_xor(t, off, 32));
      const float mn    = fmaxf(mrow[v], t);
      const float alpha = __expf(mrow[v] - mn);
      const float p0 = __expf(a0 - mn);
      const float p1 = __expf(a1 - mn);
      float rsum = p0 + p1;
#pragma unroll
      for (int off = 8; off >= 1; off >>= 1) rsum += __shfl_xor(rsum, off, 32);
      lrow[v] = lrow[v] * alpha + rsum;
      mrow[v] = mn;
#pragma unroll
      for (int dt = 0; dt < 8; ++dt) acc[dt][v] *= alpha;
      s0[v] = p0;
      s1[v] = p1;
    }

    // C-frag -> A-frag relayout via per-wave LDS scratch
    __bf16* pw = Psm + wave * (16 * PPAD);
#pragma unroll
    for (int v = 0; v < 8; ++v) {
      const int prow = v + 8 * hi;
      pw[prow * PPAD + r]      = (__bf16)s0[v];
      pw[prow * PPAD + 16 + r] = (__bf16)s1[v];
    }
    asm volatile("s_wait_dscnt 0x0" ::: "memory");
    v16bf pa;
    {
      const __bf16* p = &pw[r * PPAD + hi * 8];
      pa = cat16(*(const v8bf*)p, *(const v8bf*)(p + 16));
    }

    // Y += P(16x32) @ V(32x128)
#pragma unroll
    for (int dt = 0; dt < 8; ++dt) {
      const __bf16* p = &Vsm[(dt * 16 + r) * VPAD + hi * 8];
      acc[dt] = wmma_bf16(pa, cat16(*(const v8bf*)p, *(const v8bf*)(p + 16)), acc[dt]);
    }
    __syncthreads();
  }

  // epilogue: normalize, stage wave tile (16x128 bf16) in LDS, async store out
  __bf16* ysc = smem + wave * 2048;  // whole arena reused after final barrier
#pragma unroll
  for (int dt = 0; dt < 8; ++dt)
#pragma unroll
    for (int v = 0; v < 8; ++v)
      ysc[(v + 8 * hi) * 128 + dt * 16 + r] = (__bf16)(acc[dt][v] / lrow[v]);
  asm volatile("s_wait_dscnt 0x0" ::: "memory");

#pragma unroll
  for (int j = 0; j < 8; ++j) {          // 8 x (32 lanes x 16B) = 4KB per wave
    const int off = j * 256 + lane * 8;  // element offset in wave tile
    const int rl  = off >> 7;            // local row 0..15
    const int d   = off & 127;
    __bf16* dst = Y + ((size_t)b * T + qb + wave * 16 + rl) * C + h * D + d;
    async_store_b128(dst, ysc + off);
  }
  wait_asynccnt0();
}

// ---------------------------------------------------------------------------
extern "C" void kernel_launch(void* const* d_in, const int* in_sizes, int n_in,
                              void* d_out, int out_size, void* d_ws, size_t ws_size,
                              hipStream_t stream) {
  (void)in_sizes; (void)n_in; (void)out_size; (void)ws_size;
  const float* x      = (const float*)d_in[0];
  const float* w_attn = (const float*)d_in[1];
  const float* b_attn = (const float*)d_in[2];
  const float* w_proj = (const float*)d_in[3];
  const float* b_proj = (const float*)d_in[4];
  float* out = (float*)d_out;

  const int B = 4, T = 2048, C = 2048, H = 16;
  const int M = B * T;

  __bf16* qkv = (__bf16*)d_ws;            // [M][3C] bf16 (~100.7 MB)
  __bf16* y   = qkv + (size_t)M * 3 * C;  // [M][C]  bf16 (~33.6 MB)

  dim3 blk(256);
  gemm_bf16_wmma_kernel<float, __bf16>
      <<<dim3((3 * C) / 128, M / 128), blk, 0, stream>>>(x, w_attn, b_attn, qkv,
                                                         M, 3 * C, C);
  attn_fwd_kernel<<<dim3(B * H * (T / 128)), blk, 0, stream>>>(qkv, y, B, H, T, C);
  gemm_bf16_wmma_kernel<__bf16, float>
      <<<dim3(C / 128, M / 128), blk, 0, stream>>>(y, w_proj, b_proj, out, M, C, C);
}
